// EventSimulator_torch_59287728553986
// MI455X (gfx1250) — matrix operand order repaired and checked by reading, not compiled, not run on hardware
//
#include <hip/hip_runtime.h>
#include <hip/hip_bf16.h>

// ---------------- problem constants ----------------
#define CT 0.2f
#define KCAP 4
#define TT 48
#define HH 180
#define WW 240
#define NPIX (HH * WW)                  // 43200
#define NTOT (TT * NPIX * KCAP)         // 8,294,400 (< 2^23)
#define IPB 2048                        // items per sort block (digit counts <= 2048: f16-exact)
#define NBLK (NTOT / IPB)               // 4050 exactly
#define RADIX 256
#define NHIST (RADIX * NBLK)            // 1,036,800
#define NTILES (NHIST / 256)            // 4050 tiles of 256

typedef __attribute__((ext_vector_type(16))) _Float16 v16h;
typedef __attribute__((ext_vector_type(8)))  float    v8f;
typedef int v2i __attribute__((vector_size(8)));      // b64 payload type for async-LDS builtin

// ---- probe-guarded async global->LDS copy (ASYNCcnt path, CDNA5 ISA) ----
__device__ __forceinline__ void async_g2l_b64(const unsigned long long* g, unsigned long long* l) {
#if __has_builtin(__builtin_amdgcn_global_load_async_to_lds_b64)
    __builtin_amdgcn_global_load_async_to_lds_b64(
        (__attribute__((address_space(1))) v2i*)g,
        (__attribute__((address_space(3))) v2i*)l, 0, 0);
#else
    *l = *g;   // synchronous fallback
#endif
}
__device__ __forceinline__ void async_wait0() {
#if __has_builtin(__builtin_amdgcn_global_load_async_to_lds_b64)
#if __has_builtin(__builtin_amdgcn_s_wait_asynccnt)
    __builtin_amdgcn_s_wait_asynccnt(0);
#else
    asm volatile("s_wait_asynccnt 0x0" ::: "memory");
#endif
#endif
}

// ---------------- phase 1: per-pixel ESIM scan + packed 64-bit sort word ----------------
// word = [63]=0 | frame[62:57] | frac_bits[56:26] | slot[25:3] | polcode[2:1] | valid[0]
// Uniform timestamps => double t_ev ordering == lexicographic (frame, frac_bits); slot bits
// make every word unique, so a full 64-bit radix sort reproduces the stable argsort exactly.
__global__ __launch_bounds__(256) void esim_emit_kernel(const float* __restrict__ img,
                                                        const long long* __restrict__ ts,
                                                        unsigned long long* __restrict__ words,
                                                        int* __restrict__ tarr) {
    int p = blockIdx.x * blockDim.x + threadIdx.x;
    if (p >= NPIX) return;
    float ref     = img[p];
    float prev    = img[p];
    double tsprev = (double)ts[0];
    for (int t = 0; t < TT; ++t) {
        float cur   = img[t * NPIX + p];
        double tnow = (double)ts[t];
        float diff  = cur - ref;
        float pol   = (diff > 0.f) ? 1.f : ((diff < 0.f) ? -1.f : 0.f);
        float n     = floorf(fabsf(diff) / CT);
        float denom = cur - prev;
        unsigned long long polcode = (unsigned long long)((int)pol + 1);   // 0,1,2
        if (t + 1 < TT) __builtin_prefetch(&img[(t + 1) * NPIX + p], 0, 0); // global_prefetch_b8
#pragma unroll
        for (int k = 1; k <= KCAP; ++k) {
            float v    = ref + pol * (float)k * CT;
            float frac = (denom == 0.f) ? 0.f : ((v - prev) / denom);
            double tev = tsprev + (double)frac * (tnow - tsprev);
            bool valid = ((float)k <= n);
            int slot   = (t * NPIX + p) * KCAP + (k - 1);
            unsigned fb = __float_as_uint(frac);  // frac >= 0 for valid events -> monotone bits
            unsigned long long w;
            if (valid) {
                w = ((unsigned long long)t << 57) | ((unsigned long long)fb << 26) |
                    ((unsigned long long)slot << 3) | (polcode << 1) | 1ull;
            } else {
                w = (63ull << 57) | (0x7FFFFFFFull << 26) |
                    ((unsigned long long)slot << 3) | (polcode << 1);
            }
            words[slot] = w;
            tarr[slot]  = valid ? (int)tev : 0;
        }
        ref   += pol * n * CT;
        prev   = cur;
        tsprev = tnow;
    }
}

// ---------------- phase 2a: per-block digit histogram (digit-major), async-staged ----------------
__global__ __launch_bounds__(256) void radix_hist_kernel(const unsigned long long* __restrict__ keys,
                                                         unsigned* __restrict__ blockCounts,
                                                         int shift) {
    __shared__ unsigned long long stage[IPB];
    __shared__ unsigned cnt[RADIX];
    int t = threadIdx.x;
    cnt[t] = 0u;
    long long base = (long long)blockIdx.x * IPB;
#pragma unroll
    for (int i = 0; i < IPB / 256; ++i)
        async_g2l_b64(&keys[base + t + i * 256], &stage[t + i * 256]);
    async_wait0();
    __syncthreads();
#pragma unroll
    for (int i = 0; i < IPB / 256; ++i) {
        unsigned d = (unsigned)((stage[t + i * 256] >> shift) & 255ull);
        atomicAdd(&cnt[d], 1u);
    }
    __syncthreads();
    blockCounts[(long long)t * NBLK + blockIdx.x] = cnt[t];
}

// ---------------- phase 2b-1: parallel per-tile WMMA scan (one wave per 256-elem tile) ----
// Inclusive row prefixes of the 16x16 count matrix via v_wmma_f32_16x16x32_f16 with
// A=[M|0], B=[U;0] (U = upper-triangular ones). Counts <= 2048 are exact f16 integers;
// accumulation is f32 and totals < 2^24, so the scan is bit-exact integer arithmetic.
__global__ __launch_bounds__(256) void radix_scan_tiles(const unsigned* __restrict__ in,
                                                        unsigned* __restrict__ outLocal,
                                                        unsigned* __restrict__ tileTotals,
                                                        int ntiles) {
    __shared__ float lcnt[8][256];
    __shared__ float rbuf[8][256];
    __shared__ float rowoff[8][16];
    int wv = threadIdx.x >> 5;
    int l  = threadIdx.x & 31;
    int tile = blockIdx.x * 8 + wv;
    bool live = (tile < ntiles);
    int tileC = live ? tile : 0;
    long long base = (long long)tileC * 256;

    int nn = l & 15;
    bool hi = (l >= 16);
    v16h bU;
#pragma unroll
    for (int h = 0; h < 16; ++h)
        bU[h] = (_Float16)(((h + ((h < 8) ? 0 : 8) + (hi ? 8 : 0)) < 16 && ((h + ((h < 8) ? 0 : 8) + (hi ? 8 : 0)) <= nn)) ? 1.0f : 0.0f);

#pragma unroll
    for (int j = 0; j < 8; ++j) lcnt[wv][l * 8 + j] = (float)in[base + l * 8 + j];
    __syncthreads();
    int m = l & 15;
    v16h a;
#pragma unroll
    for (int h = 0; h < 16; ++h) {
        int k = h + ((h < 8) ? 0 : 8) + (hi ? 8 : 0);
        a[h] = (_Float16)((k < 16) ? lcnt[wv][m * 16 + k] : 0.0f);
    }
    v8f c = {};
    c = __builtin_amdgcn_wmma_f32_16x16x32_f16(false, a, false, bU, (short)0, c, false, false);
#pragma unroll
    for (int j = 0; j < 8; ++j) {
        int mm = j + (hi ? 8 : 0);
        rbuf[wv][mm * 16 + nn] = c[j];
    }
    __syncthreads();
    if (l == 0) {
        float acc = 0.f;
        for (int r = 0; r < 16; ++r) { rowoff[wv][r] = acc; acc += rbuf[wv][r * 16 + 15]; }
        if (live) tileTotals[tileC] = (unsigned)acc;
    }
    __syncthreads();
    if (live) {
#pragma unroll
        for (int j = 0; j < 8; ++j) {
            int idx = l * 8 + j;
            int mm  = idx >> 4;
            outLocal[base + idx] = (unsigned)(rowoff[wv][mm] + rbuf[wv][idx] - lcnt[wv][idx]);
        }
    }
}

// ---------------- phase 2b-2: exact u32 scan of the 4050 tile totals (single wave) ----------
__global__ __launch_bounds__(32) void radix_scan_totals(const unsigned* __restrict__ totals,
                                                        unsigned* __restrict__ prefix,
                                                        int n) {
    int l = threadIdx.x;
    unsigned carry = 0u;
    for (int base = 0; base < n; base += 32) {
        int idx = base + l;
        unsigned v = (idx < n) ? totals[idx] : 0u;
        int s = (int)v;
#pragma unroll
        for (int d = 1; d < 32; d <<= 1) {
            int t = __shfl_up(s, d, 32);
            if (l >= d) s += t;
        }
        if (idx < n) prefix[idx] = (unsigned)s - v + carry;
        carry += (unsigned)__shfl((int)s, 31, 32);
    }
}

// ---------------- phase 2c: scatter (offset = in-tile scan + tile prefix) ----------------
__global__ __launch_bounds__(256) void radix_scatter_kernel(const unsigned long long* __restrict__ keysIn,
                                                            unsigned long long* __restrict__ keysOut,
                                                            const unsigned* __restrict__ localOff,
                                                            const unsigned* __restrict__ tilePref,
                                                            int shift) {
    __shared__ unsigned cnt[RADIX];
    int t = threadIdx.x;
    cnt[t] = 0u;
    __syncthreads();
    long long base = (long long)blockIdx.x * IPB;
#pragma unroll
    for (int i = 0; i < IPB / 256; ++i) {
        long long idx = base + t + i * 256;
        unsigned long long k = keysIn[idx];
        unsigned d    = (unsigned)((k >> shift) & 255ull);
        unsigned rank = atomicAdd(&cnt[d], 1u);
        unsigned e    = d * (unsigned)NBLK + blockIdx.x;
        unsigned pos  = localOff[e] + tilePref[e >> 8] + rank;
        keysOut[pos]  = k;
    }
}

// ---------------- phase 3: decode sorted words into 5 output arrays ----------------
__global__ __launch_bounds__(256) void finalize_kernel(const unsigned long long* __restrict__ words,
                                                       const int* __restrict__ tarr,
                                                       float* __restrict__ out) {
    long long i = (long long)blockIdx.x * blockDim.x + threadIdx.x;
    if (i >= (long long)NTOT) return;
    unsigned long long w = words[i];
    int slot  = (int)((w >> 3) & 0x7FFFFFull);
    int pol   = (int)((w >> 1) & 3ull) - 1;
    int valid = (int)(w & 1ull);
    int tp    = slot >> 2;           // t*NPIX + pixel
    int pix   = tp % NPIX;
    out[i]                   = (float)(pix % WW);
    out[(long long)NTOT + i] = (float)(pix / WW);
    out[2LL * NTOT + i]      = (float)tarr[slot];
    out[3LL * NTOT + i]      = (float)pol;
    out[4LL * NTOT + i]      = (float)valid;
}

// ---------------- launch ----------------
extern "C" void kernel_launch(void* const* d_in, const int* in_sizes, int n_in,
                              void* d_out, int out_size, void* d_ws, size_t ws_size,
                              hipStream_t stream) {
    const float* img    = (const float*)d_in[0];
    const long long* ts = (const long long*)d_in[1];   // int64 timestamps
    float* out          = (float*)d_out;

    char* ws = (char*)d_ws;
    size_t off = 0;
    auto carve = [&](size_t bytes) -> void* {
        void* p = ws + off;
        off += (bytes + 255) & ~(size_t)255;
        return p;
    };
    unsigned long long* wordsA = (unsigned long long*)carve((size_t)NTOT * 8);
    unsigned long long* wordsB = (unsigned long long*)carve((size_t)NTOT * 8);
    int* tarr                  = (int*)carve((size_t)NTOT * 4);
    unsigned* blockCounts      = (unsigned*)carve((size_t)NHIST * 4);
    unsigned* localOff         = (unsigned*)carve((size_t)NHIST * 4);
    unsigned* tileTotals       = (unsigned*)carve((size_t)NTILES * 4);
    unsigned* tilePref         = (unsigned*)carve((size_t)NTILES * 4);

    // 1) emit packed sort words + truncated timestamps
    esim_emit_kernel<<<(NPIX + 255) / 256, 256, 0, stream>>>(img, ts, wordsA, tarr);

    // 2) 8-pass LSD radix sort on the full 64-bit word (exactly reproduces stable argsort)
    unsigned long long* kin = wordsA; unsigned long long* kout = wordsB;
    for (int pass = 0; pass < 8; ++pass) {
        int shift = pass * 8;
        radix_hist_kernel<<<NBLK, 256, 0, stream>>>(kin, blockCounts, shift);
        radix_scan_tiles<<<(NTILES + 7) / 8, 256, 0, stream>>>(blockCounts, localOff, tileTotals, NTILES);
        radix_scan_totals<<<1, 32, 0, stream>>>(tileTotals, tilePref, NTILES);
        radix_scatter_kernel<<<NBLK, 256, 0, stream>>>(kin, kout, localOff, tilePref, shift);
        unsigned long long* tk = kin; kin = kout; kout = tk;
    }

    // 3) decode into the 5 concatenated float output arrays
    finalize_kernel<<<(NTOT + 255) / 256, 256, 0, stream>>>(kin, tarr, out);
}
